// HomomorphicVocoder_71433896067429
// MI455X (gfx1250) — compile-verified, matrix operand order checked by reading, not run on hardware
//
#include <hip/hip_runtime.h>
#include <math.h>

// ---------- sizes ----------
#define Bn      8
#define Tn      1200
#define Fn      1201          // STFT frames (T+1)
#define Ln      307200        // audio samples
#define NFFT    1024
#define HOP     256
#define BINS    513
#define SRf     24000.0f
#define MROW    9616          // B*F=9608 padded to 16
#define LDSPEC  1088          // Re 0..512 | pad | Im at +544 .. 1056 | pad (68 n-tiles)
#define IMOFF   544
#define LDLOG   544           // 513 log-amp cols padded to 32-multiple
#define OLAL    308224        // L + NFFT
#define KREV    8192
#define KTOEP   8224          // 514*16, 32-multiple
#define V16PAD  8224          // left zero pad (>= 513 blocks * 16)
#define V16S    (V16PAD + Ln) // padded f16 signal row stride = 315424
#define TWO_PI_OVER_N 0.006135923151542565f

typedef _Float16     v16h  __attribute__((ext_vector_type(16)));
typedef float        v8f   __attribute__((ext_vector_type(8)));
typedef unsigned int u32x4 __attribute__((ext_vector_type(4)));
typedef int          i32x8 __attribute__((ext_vector_type(8)));
typedef int          i32x4 __attribute__((ext_vector_type(4)));

__device__ __forceinline__ float ang(int prod) {
    return TWO_PI_OVER_N * (float)(prod & (NFFT - 1));
}
__device__ __forceinline__ float liftf(int n) {
    return (n == 0 || n == 512) ? 1.0f : (n < 512 ? 2.0f : 0.0f);
}
__device__ __forceinline__ float wkf(int k) {
    return (k == 0 || k == 512) ? 1.0f : 2.0f;
}

// ---------- matrix builders (ALL stored transposed: [N][K]) ----------
__global__ void k_build_dfwdT(_Float16* D) {           // [1088][1024]: row=out col, col=n
    long idx = (long)blockIdx.x * blockDim.x + threadIdx.x;
    if (idx >= (long)LDSPEC * NFFT) return;
    int col = (int)(idx / NFFT), n = (int)(idx % NFFT);
    float v = 0.0f;
    if (col < BINS)                              v =  __cosf(ang(n * col));
    else if (col >= IMOFF && col < IMOFF + BINS) v = -__sinf(ang(n * (col - IMOFF)));
    D[idx] = (_Float16)v;
}
__global__ void k_build_ccepT(_Float16* C) {           // [1024][544]: row=n, col=k
    long idx = (long)blockIdx.x * blockDim.x + threadIdx.x;
    if (idx >= (long)NFFT * LDLOG) return;
    int n = (int)(idx / LDLOG), k = (int)(idx % LDLOG);
    float v = (k <= 512) ? wkf(k) * (1.0f / 1024.0f) * __cosf(ang(k * n)) : 0.0f;
    C[idx] = (_Float16)v;
}
__global__ void k_build_ematT(_Float16* E) {           // [1088][1024]: row=out col, col=n
    long idx = (long)blockIdx.x * blockDim.x + threadIdx.x;
    if (idx >= (long)LDSPEC * NFFT) return;
    int col = (int)(idx / NFFT), n = (int)(idx % NFFT);
    float lf = liftf(n), v = 0.0f;
    if (col < BINS)                              v =  lf * __cosf(ang(n * col));
    else if (col >= IMOFF && col < IMOFF + BINS) v = -lf * __sinf(ang(n * (col - IMOFF)));
    E[idx] = (_Float16)v;
}
__global__ void k_build_dinvT(_Float16* D) {           // [1024][1088]: row=time n, col=spec row
    long idx = (long)blockIdx.x * blockDim.x + threadIdx.x;
    if (idx >= (long)NFFT * LDSPEC) return;
    int n = (int)(idx / LDSPEC), row = (int)(idx % LDSPEC);
    float v = 0.0f;
    if (row < BINS)                              v =  wkf(row) * (1.0f / 1024.0f) * __cosf(ang(row * n));
    else if (row >= IMOFF && row < IMOFF + BINS) {
        int k = row - IMOFF;
        v = -wkf(k) * (1.0f / 1024.0f) * __sinf(ang(k * n));
    }
    D[idx] = (_Float16)v;
}

// ---------- impulse train: f0 upsample + block-wise phase scan ----------
__global__ void k_impulse(const float* __restrict__ f0, float* __restrict__ imp) {
    int b = blockIdx.x, tid = threadIdx.x;                 // 256 threads
    __shared__ float sf0[Tn];
    __shared__ float s[256];
    __shared__ float carry;
    for (int i = tid; i < Tn; i += 256) sf0[i] = f0[(long)b * Tn + i];
    if (tid == 0) carry = 0.0f;
    __syncthreads();
    const float sscale = sqrtf(SRf);
    for (int base = 0; base < Ln; base += 256) {
        int t = base + tid;
        float src = (t + 0.5f) * (1.0f / HOP) - 0.5f;
        src = fminf(fmaxf(src, 0.0f), (float)(Tn - 1));
        int i0 = (int)floorf(src);
        int i1 = min(i0 + 1, Tn - 1);
        float w = src - (float)i0;
        float v = sf0[i0] * (1.0f - w) + sf0[i1] * w;      // f0_up
        float ph = v * (1.0f / SRf);
        s[tid] = ph; __syncthreads();
        for (int off = 1; off < 256; off <<= 1) {          // inclusive scan
            float t2 = (tid >= off) ? s[tid - off] : 0.0f;
            __syncthreads();
            s[tid] += t2;
            __syncthreads();
        }
        float phase = carry + s[tid];
        float sawp  = fmodf(phase - ph, 1.0f);
        float saw   = fmodf(phase, 1.0f);
        float val = 0.0f;
        if (t > 0 && (saw - sawp) < 0.0f)
            val = rsqrtf(fmaxf(v, 20.0f)) * sscale;
        imp[(long)b * Ln + t] = val;
        __syncthreads();
        if (tid == 255) carry = phase;
        __syncthreads();
    }
}

// ---------- framing (reflect pad + window) -> f16 ----------
__global__ void k_frames(const float* __restrict__ sig, const float* __restrict__ win,
                         _Float16* __restrict__ out) {
    long idx = (long)blockIdx.x * blockDim.x + threadIdx.x;
    if (idx >= (long)MROW * NFFT) return;
    int n = (int)(idx & (NFFT - 1));
    long row = idx >> 10;
    float v = 0.0f;
    if (row < (long)Bn * Fn) {
        int b = (int)(row / Fn), f = (int)(row % Fn);
        int s = f * HOP + n - NFFT / 2;
        if (s < 0) s = -s;
        if (s >= Ln) s = 2 * Ln - 2 - s;
        v = sig[(long)b * Ln + s] * win[n];
    }
    out[idx] = (_Float16)v;
}

// ---------- WMMA GEMM, 16x64 strip per wave: C[f32] = A[MxK] * BT[NxK]^T ----------
__global__ void k_gemm(const _Float16* __restrict__ A, int lda,
                       const _Float16* __restrict__ BT, int ldk,
                       float* __restrict__ C, int ldc,
                       int tiles_m, int strips_n, int K) {
    int wid  = (int)(((long)blockIdx.x * blockDim.x + threadIdx.x) >> 5);
    int lane = threadIdx.x & 31;
    if (wid >= tiles_m * strips_n) return;                 // wave-uniform: EXEC stays full
    int tm = wid / strips_n, ts = wid % strips_n;
    int i  = lane & 15, half = lane >> 4;
    const _Float16* arow = A  + (long)(tm * 16 + i) * lda + (half << 4);
    const _Float16* b0   = BT + (long)(ts * 64 + i) * ldk + (half << 4);
    v8f c0 = {}, c1 = {}, c2 = {}, c3 = {};
    for (int k0 = 0; k0 < K; k0 += 32) {
        __builtin_prefetch(arow + k0 + 256, 0, 1);
        v16h a   = *(const v16h*)(arow + k0);
        v16h bf0 = *(const v16h*)(b0 + k0);
        v16h bf1 = *(const v16h*)(b0 + 16L * ldk + k0);
        v16h bf2 = *(const v16h*)(b0 + 32L * ldk + k0);
        v16h bf3 = *(const v16h*)(b0 + 48L * ldk + k0);
        c0 = __builtin_amdgcn_wmma_f32_16x16x32_f16(false, a, false, bf0, (short)0, c0, false, false);
        c1 = __builtin_amdgcn_wmma_f32_16x16x32_f16(false, a, false, bf1, (short)0, c1, false, false);
        c2 = __builtin_amdgcn_wmma_f32_16x16x32_f16(false, a, false, bf2, (short)0, c2, false, false);
        c3 = __builtin_amdgcn_wmma_f32_16x16x32_f16(false, a, false, bf3, (short)0, c3, false, false);
    }
    float* crow = C + (long)(tm * 16 + 8 * half) * ldc + (ts * 64 + i);
#pragma unroll
    for (int r = 0; r < 8; ++r) {
        crow[(long)r * ldc +  0] = c0[r];
        crow[(long)r * ldc + 16] = c1[r];
        crow[(long)r * ldc + 32] = c2[r];
        crow[(long)r * ldc + 48] = c3[r];
    }
}

// ---------- log-amplitude of gated periodic envelope ----------
__global__ void k_logamp(const float* __restrict__ env_per, const float* __restrict__ f0,
                         _Float16* __restrict__ out) {
    long idx = (long)blockIdx.x * blockDim.x + threadIdx.x;
    if (idx >= (long)MROW * LDLOG) return;
    long row = idx / LDLOG; int k = (int)(idx % LDLOG);
    float v = 0.0f;
    if (row < (long)Bn * Fn && k <= 512) {
        int b = (int)(row / Fn), f = (int)(row % Fn);
        float env = 0.0f;
        if (f > 0) {
            env = env_per[((long)b * BINS + k) * Tn + (f - 1)];
            if (!(f0[(long)b * Tn + (f - 1)] > 20.0f)) env = 0.0f;
        }
        v = __logf(fmaxf(env, 1e-5f));
    }
    out[idx] = (_Float16)v;
}

__global__ void k_cvt_f32_f16(const float* __restrict__ in, _Float16* __restrict__ out, long n) {
    long idx = (long)blockIdx.x * blockDim.x + threadIdx.x;
    if (idx < n) out[idx] = (_Float16)in[idx];
}

// ---------- min-phase exp + complex mix -> voi spectrum (f16) ----------
__global__ void k_combine(const float* __restrict__ logspec,
                          const float* __restrict__ sA,  // imp stft
                          const float* __restrict__ sB,  // noi stft
                          const float* __restrict__ env_noi,
                          _Float16* __restrict__ vsp) {
    long idx = (long)blockIdx.x * blockDim.x + threadIdx.x;
    if (idx >= (long)MROW * LDSPEC) return;
    long row = idx / LDSPEC; int col = (int)(idx % LDSPEC);
    float v = 0.0f;
    int k = -1, isRe = 1;
    if (col < BINS) { k = col; isRe = 1; }
    else if (col >= IMOFF && col < IMOFF + BINS) { k = col - IMOFF; isRe = 0; }
    if (row < (long)Bn * Fn && k >= 0) {
        int b = (int)(row / Fn), f = (int)(row % Fn);
        long rb = row * LDSPEC;
        float lr = logspec[rb + k], li = logspec[rb + IMOFF + k];
        float m  = __expf(lr);
        float pre = m * __cosf(li), pim = m * __sinf(li);
        float ir = sA[rb + k], ii = sA[rb + IMOFF + k];
        float nr = sB[rb + k], ni = sB[rb + IMOFF + k];
        float en = (f > 0) ? env_noi[((long)b * BINS + k) * Tn + (f - 1)] : 0.0f;
        float vr = ir * pre - ii * pim + nr * en;
        float vi = ir * pim + ii * pre + ni * en;
        v = isRe ? vr : vi;
    }
    vsp[idx] = (_Float16)v;
}

// ---------- overlap-add ----------
__global__ void k_fill_zero(float* p, long n) {
    long idx = (long)blockIdx.x * blockDim.x + threadIdx.x;
    if (idx < n) p[idx] = 0.0f;
}
__global__ void k_ola(const float* __restrict__ frames, const float* __restrict__ win,
                      float* __restrict__ ola) {
    long idx = (long)blockIdx.x * blockDim.x + threadIdx.x;
    if (idx >= (long)Bn * Fn * NFFT) return;
    int n = (int)(idx & (NFFT - 1));
    long row = idx >> 10;
    int b = (int)(row / Fn), f = (int)(row % Fn);
    float v = frames[row * NFFT + n] * win[n];
    atomicAdd(&ola[(long)b * OLAL + (long)f * HOP + n], v);
}
__global__ void k_norm_v16(const float* __restrict__ ola, const float* __restrict__ win,
                           _Float16* __restrict__ v16) {
    long idx = (long)blockIdx.x * blockDim.x + threadIdx.x;
    if (idx >= (long)Bn * V16S) return;
    int b = (int)(idx / V16S);
    int j = (int)(idx % V16S);
    float v = 0.0f;
    if (j >= V16PAD) {
        int t = j - V16PAD;                 // 0..L-1
        int s = t + NFFT / 2;               // padded-domain sample
        float wsq = 0.0f;
        int nb = s & (HOP - 1);
#pragma unroll
        for (int jj = 0; jj < 4; ++jj) {
            int n = nb + HOP * jj;
            int f = (s - n) >> 8;
            if (f >= 0 && f < Fn) { float w = win[n]; wsq += w * w; }
        }
        v = ola[(long)b * OLAL + s] / (wsq > 1e-11f ? wsq : 1.0f);
    }
    v16[idx] = (_Float16)v;
}

// ---------- banded-Toeplitz kernel matrix, TRANSPOSED [16][KTOEP] per batch ----------
__global__ void k_btoep(const float* __restrict__ rev, _Float16* __restrict__ Bt) {
    long idx = (long)blockIdx.x * blockDim.x + threadIdx.x;
    if (idx >= (long)Bn * 16 * KTOEP) return;
    int k = (int)(idx % KTOEP);
    int n = (int)((idx / KTOEP) % 16);
    int b = (int)(idx / ((long)16 * KTOEP));
    int q = k >> 4, r = k & 15;
    int j = n - r + 16 * q;
    float v = (j >= 0 && j < KREV) ? rev[(long)b * KREV + j] : 0.0f;
    Bt[idx] = (_Float16)v;
}

// ---------- 8192-tap reverb as per-batch WMMA GEMM ----------
// B staged once per workgroup into 263KB LDS via the Tensor Data Mover (TDM),
// 4 M-tiles per wave reuse each LDS B fragment.
__global__ void __launch_bounds__(128)
k_conv(const _Float16* __restrict__ v16, const _Float16* __restrict__ Bt,
       float* __restrict__ out) {
    extern __shared__ _Float16 ldsB[];                     // [16][KTOEP] = 263168 B
    int b    = blockIdx.x / 75;                            // 75 blocks * 4 waves * 4 tiles = 1200
    int grp  = blockIdx.x % 75;
    int wave = threadIdx.x >> 5;
    int lane = threadIdx.x & 31;

#if __has_builtin(__builtin_amdgcn_tensor_load_to_lds) && __has_builtin(__builtin_amdgcn_s_wait_tensorcnt)
    if (wave == 0) {
        // D# group0: count=1 | lds_addr | 57b global addr | type=2
        unsigned int laddr = (unsigned int)(unsigned long)&ldsB[0];
        unsigned long gaddr = (unsigned long)(Bt + (long)b * 16 * KTOEP);
        u32x4 g0;
        g0[0] = 1u;
        g0[1] = laddr;
        g0[2] = (unsigned int)(gaddr & 0xFFFFFFFFu);
        g0[3] = (unsigned int)((gaddr >> 32) & 0x01FFFFFFu) | (2u << 30);
        // D# group1: 2D tile, data_size=8B, 32896 elements (263168 B), 1 row
        const unsigned int NE = (16u * KTOEP * 2u) / 8u;   // 32896
        i32x8 g1;
        g1[0] = (int)(3u << 16);                           // wg_mask=0, data_size=8B
        g1[1] = (int)((NE & 0xFFFFu) << 16);               // tensor_dim0[15:0]  @ bits 63:48
        g1[2] = (int)((NE >> 16) | (1u << 16));            // tensor_dim0[31:16], tensor_dim1=1
        g1[3] = (int)((NE & 0xFFFFu) << 16);               // tile_dim0 @ bits 127:112
        g1[4] = (int)1;                                    // tile_dim1 = 1
        g1[5] = (int)NE;                                   // tensor_dim0_stride[31:0]
        g1[6] = 0; g1[7] = 0;
        i32x4 g2 = {0, 0, 0, 0};                           // groups 2/3 unused (<=2D tile)
        i32x4 g3 = {0, 0, 0, 0};
        i32x8 g4 = {0, 0, 0, 0, 0, 0, 0, 0};               // 6-arg (clang-23) extra group
        __builtin_amdgcn_tensor_load_to_lds(g0, g1, g2, g3, g4, 0);
        __builtin_amdgcn_s_wait_tensorcnt(0);
    }
#else
    {
        const uint4* src = (const uint4*)(Bt + (long)b * 16 * KTOEP);
        uint4* dst = (uint4*)ldsB;
        for (int t = threadIdx.x; t < (16 * KTOEP) / 8; t += 128) dst[t] = src[t];
    }
#endif
    __syncthreads();

    int i = lane & 15, half = lane >> 4;
    const _Float16* arow = v16 + (long)b * V16S + V16PAD;
    int tmb = (grp * 4 + wave) * 4;                        // 4 consecutive M-tiles
    v8f c0 = {}, c1 = {}, c2 = {}, c3 = {};
    for (int kq = 0; kq < KTOEP / 32; ++kq) {              // 257 chunks of K=32
        int q  = 2 * kq + half;
        int kk = 32 * kq + half * 16;
        v16h bf = *(const v16h*)&ldsB[(long)i * KTOEP + kk];
        const _Float16* ap = arow + 16L * ((tmb * 16) + i - q);
        __builtin_prefetch(ap + 1024, 0, 1);
        v16h a0 = *(const v16h*)(ap);
        v16h a1 = *(const v16h*)(ap + 16L * 16);
        v16h a2 = *(const v16h*)(ap + 32L * 16);
        v16h a3 = *(const v16h*)(ap + 48L * 16);
        c0 = __builtin_amdgcn_wmma_f32_16x16x32_f16(false, a0, false, bf, (short)0, c0, false, false);
        c1 = __builtin_amdgcn_wmma_f32_16x16x32_f16(false, a1, false, bf, (short)0, c1, false, false);
        c2 = __builtin_amdgcn_wmma_f32_16x16x32_f16(false, a2, false, bf, (short)0, c2, false, false);
        c3 = __builtin_amdgcn_wmma_f32_16x16x32_f16(false, a3, false, bf, (short)0, c3, false, false);
    }
#pragma unroll
    for (int t = 0; t < 4; ++t) {
        int m0 = (tmb + t) * 16;
        float* op = out + (long)b * Ln + 16L * (m0 + 8 * half) + i;
        v8f c = (t == 0) ? c0 : (t == 1) ? c1 : (t == 2) ? c2 : c3;
#pragma unroll
        for (int r = 0; r < 8; ++r) op[16L * r] = c[r];
    }
}

// ======================= host-side launch =======================
extern "C" void kernel_launch(void* const* d_in, const int* in_sizes, int n_in,
                              void* d_out, int out_size, void* d_ws, size_t ws_size,
                              hipStream_t stream) {
    (void)in_sizes; (void)n_in; (void)out_size; (void)ws_size;
    const float* f0      = (const float*)d_in[0];
    const float* env_per = (const float*)d_in[1];
    const float* env_noi = (const float*)d_in[2];
    const float* rev     = (const float*)d_in[3];
    const float* noi     = (const float*)d_in[4];
    const float* win     = (const float*)d_in[5];
    float* out = (float*)d_out;

    char* ws = (char*)d_ws;
    size_t off = 0;
    auto take = [&](size_t bytes) -> char* {
        char* p = ws + off;
        off = (off + bytes + 255) & ~(size_t)255;
        return p;
    };
    float*    imp    = (float*)   take((size_t)Bn * Ln * 4);
    _Float16* frImp  = (_Float16*)take((size_t)MROW * NFFT * 2);
    _Float16* frNoi  = (_Float16*)take((size_t)MROW * NFFT * 2);
    _Float16* DfwdT  = (_Float16*)take((size_t)LDSPEC * NFFT * 2);
    float*    sImp   = (float*)   take((size_t)MROW * LDSPEC * 4);
    float*    sNoi   = (float*)   take((size_t)MROW * LDSPEC * 4);
    _Float16* logA   = (_Float16*)take((size_t)MROW * LDLOG * 2);
    _Float16* CcepT  = (_Float16*)take((size_t)NFFT * LDLOG * 2);
    float*    cepF   = (float*)   take((size_t)MROW * NFFT * 4);   // reused as framesout
    _Float16* cepH   = (_Float16*)take((size_t)MROW * NFFT * 2);
    _Float16* EmatT  = (_Float16*)take((size_t)LDSPEC * NFFT * 2);
    float*    logS   = (float*)   take((size_t)MROW * LDSPEC * 4);
    _Float16* vsp    = (_Float16*)take((size_t)MROW * LDSPEC * 2);
    _Float16* DinvT  = (_Float16*)take((size_t)NFFT * LDSPEC * 2);
    float*    ola    = (float*)   take((size_t)Bn * OLAL * 4);
    _Float16* v16    = (_Float16*)take((size_t)Bn * V16S * 2);
    _Float16* Bt     = (_Float16*)take((size_t)Bn * 16 * KTOEP * 2);

    auto cdiv = [](long a, long b) { return (int)((a + b - 1) / b); };
    const int TPB = 256;

    // transposed operand matrices
    k_build_dfwdT<<<cdiv((long)LDSPEC * NFFT, TPB), TPB, 0, stream>>>(DfwdT);
    k_build_ccepT<<<cdiv((long)NFFT * LDLOG, TPB), TPB, 0, stream>>>(CcepT);
    k_build_ematT<<<cdiv((long)LDSPEC * NFFT, TPB), TPB, 0, stream>>>(EmatT);
    k_build_dinvT<<<cdiv((long)NFFT * LDSPEC, TPB), TPB, 0, stream>>>(DinvT);

    // impulse train
    k_impulse<<<Bn, TPB, 0, stream>>>(f0, imp);

    // framing
    k_frames<<<cdiv((long)MROW * NFFT, TPB), TPB, 0, stream>>>(imp, win, frImp);
    k_frames<<<cdiv((long)MROW * NFFT, TPB), TPB, 0, stream>>>(noi, win, frNoi);

    const int TM = MROW / 16;           // 601
    const int SN_SPEC = LDSPEC / 64;    // 17
    const int SN_TIME = NFFT / 64;      // 16

    // STFTs: [9616x1024] x [1024x1088]
    k_gemm<<<cdiv((long)TM * SN_SPEC, 8), TPB, 0, stream>>>(frImp, NFFT, DfwdT, NFFT, sImp, LDSPEC, TM, SN_SPEC, NFFT);
    k_gemm<<<cdiv((long)TM * SN_SPEC, 8), TPB, 0, stream>>>(frNoi, NFFT, DfwdT, NFFT, sNoi, LDSPEC, TM, SN_SPEC, NFFT);

    // min-phase: log-amp -> cepstrum GEMM -> lifted rfft GEMM
    k_logamp<<<cdiv((long)MROW * LDLOG, TPB), TPB, 0, stream>>>(env_per, f0, logA);
    k_gemm<<<cdiv((long)TM * SN_TIME, 8), TPB, 0, stream>>>(logA, LDLOG, CcepT, LDLOG, cepF, NFFT, TM, SN_TIME, LDLOG);
    k_cvt_f32_f16<<<cdiv((long)MROW * NFFT, TPB), TPB, 0, stream>>>(cepF, cepH, (long)MROW * NFFT);
    k_gemm<<<cdiv((long)TM * SN_SPEC, 8), TPB, 0, stream>>>(cepH, NFFT, EmatT, NFFT, logS, LDSPEC, TM, SN_SPEC, NFFT);

    // complex mix -> voi spectrum
    k_combine<<<cdiv((long)MROW * LDSPEC, TPB), TPB, 0, stream>>>(logS, sImp, sNoi, env_noi, vsp);

    // iSTFT: [9616x1088] x [1088x1024] -> time frames (reuse cepF)
    k_gemm<<<cdiv((long)TM * SN_TIME, 8), TPB, 0, stream>>>(vsp, LDSPEC, DinvT, LDSPEC, cepF, NFFT, TM, SN_TIME, LDSPEC);

    // overlap-add + normalize + f16 padded signal
    k_fill_zero<<<cdiv((long)Bn * OLAL, TPB), TPB, 0, stream>>>(ola, (long)Bn * OLAL);
    k_ola<<<cdiv((long)Bn * Fn * NFFT, TPB), TPB, 0, stream>>>(cepF, win, ola);
    k_norm_v16<<<cdiv((long)Bn * V16S, TPB), TPB, 0, stream>>>(ola, win, v16);

    // reverb: transposed Toeplitz kernel matrix + per-batch WMMA conv GEMM
    k_btoep<<<cdiv((long)Bn * 16 * KTOEP, TPB), TPB, 0, stream>>>(rev, Bt);
    k_conv<<<Bn * 75, 128, (size_t)16 * KTOEP * 2, stream>>>(v16, Bt, out);
}
